// RelMultiHeadAttn_71442486002123
// MI455X (gfx1250) — matrix operand order repaired and checked by reading, not compiled, not run on hardware
//
#include <hip/hip_runtime.h>

#define QLEN   2048
#define BSZ    4
#define NHEAD  8
#define DHEAD  64
#define DMODEL 512

typedef __attribute__((ext_vector_type(16))) _Float16 v16h;
typedef __attribute__((ext_vector_type(8)))  _Float16 v8h;
typedef __attribute__((ext_vector_type(8)))  float    v8f;
typedef __attribute__((ext_vector_type(4)))  unsigned int u32x4;
typedef __attribute__((ext_vector_type(8)))  int      i32x8;
typedef __attribute__((ext_vector_type(4)))  int      i32x4;

// Tensor Data Mover availability (guarded: ROCm7.2 = 5-arg builtin,
// therock/amdgpu-toolchain = 6-arg builtin + TDM header).
#if __has_builtin(__builtin_amdgcn_tensor_load_to_lds) && \
    __has_builtin(__builtin_amdgcn_s_wait_tensorcnt)
#define HAVE_TDM 1
#if __has_include(<hip/amd_detail/amd_gfx1250_TDM.h>)
#define TDM_SIX_ARGS 1
#else
#define TDM_SIX_ARGS 0
#endif
#else
#define HAVE_TDM 0
#endif

#define WMMA_F16(a, b, c) \
  __builtin_amdgcn_wmma_f32_16x16x32_f16(false, (a), false, (b), (short)0, (c), false, false)

// ---------------------------------------------------------------------------
// Fragment helpers (gfx1250 wave32 WMMA; A/B use this file's contiguous k order, D per ISA)
// A 16x32 f16: lane m=lane&15, hh=lane>>4; halves 0..15 = A[m][k0+hh*16 .. +15]
// B 32x16 f16: lane col n=lane&15; halves = Bmath[k0+hh*16 .. +15][n]
// D 16x16 f32: lane col n=lane&15; VGPR v -> row v + 8*hh
// ---------------------------------------------------------------------------
__device__ __forceinline__ v16h pack16(v8h lo, v8h hi) {
  v16h a;
#pragma unroll
  for (int t = 0; t < 8; ++t) { a[t] = lo[t]; a[t + 8] = hi[t]; }
  return a;
}

__device__ __forceinline__ v16h load_afrag(const _Float16* row, int k0, int hh) {
  v8h lo = *(const v8h*)(row + k0 + hh * 16);
  v8h hi = *(const v8h*)(row + k0 + hh * 16 + 8);
  return pack16(lo, hi);
}

// ---------------------------------------------------------------------------
// Kernel 1: pre-LayerNorm + f16 cast.  One block (128 thr) per row of 512.
// ---------------------------------------------------------------------------
__global__ __launch_bounds__(128) void ln_cast_kernel(const float* __restrict__ w,
                                                      const float* __restrict__ gamma,
                                                      const float* __restrict__ beta,
                                                      _Float16* __restrict__ wn) {
  __shared__ float s1[4], s2[4];
  const int r = blockIdx.x;
  const float* x = w + (size_t)r * DMODEL;
  float4 xv = ((const float4*)x)[threadIdx.x];
  float s = xv.x + xv.y + xv.z + xv.w;
  float q = xv.x * xv.x + xv.y * xv.y + xv.z * xv.z + xv.w * xv.w;
#pragma unroll
  for (int msk = 1; msk < 32; msk <<= 1) {
    s += __shfl_xor(s, msk, 32);
    q += __shfl_xor(q, msk, 32);
  }
  const int wv = threadIdx.x >> 5;
  if ((threadIdx.x & 31) == 0) { s1[wv] = s; s2[wv] = q; }
  __syncthreads();
  float S = s1[0] + s1[1] + s1[2] + s1[3];
  float Q = s2[0] + s2[1] + s2[2] + s2[3];
  float mu = S * (1.0f / DMODEL);
  float var = Q * (1.0f / DMODEL) - mu * mu;
  float rstd = rsqrtf(var + 1e-5f);
  const int c = threadIdx.x * 4;
  float xe[4] = {xv.x, xv.y, xv.z, xv.w};
#pragma unroll
  for (int k = 0; k < 4; ++k)
    wn[(size_t)r * DMODEL + c + k] =
        (_Float16)((xe[k] - mu) * rstd * gamma[c + k] + beta[c + k]);
}

// ---------------------------------------------------------------------------
// Kernel 2: sinusoidal relative position embedding -> f16 [QLEN, DMODEL]
// ---------------------------------------------------------------------------
__global__ __launch_bounds__(128) void posemb_kernel(_Float16* __restrict__ rh) {
  const int j = blockIdx.x;
  const float pos = (float)(QLEN - 1 - j);
  const float lg = logf(10000.0f) / 256.0f;
#pragma unroll
  for (int k = 0; k < 4; ++k) {
    int cc = threadIdx.x * 4 + k;
    int i = cc & 255;
    float invf = __expf(-(float)i * lg);
    float ang = pos * invf;
    float val = (cc < 256) ? sinf(ang) : cosf(ang);
    rh[(size_t)j * DMODEL + cc] = (_Float16)val;
  }
}

// ---------------------------------------------------------------------------
// Kernel 3: f32 -> f16 cast (weights)
// ---------------------------------------------------------------------------
__global__ void cast_f16_kernel(const float* __restrict__ s, _Float16* __restrict__ d, int n) {
  int i = blockIdx.x * blockDim.x + threadIdx.x;
  if (i < n) d[i] = (_Float16)s[i];
}

// ---------------------------------------------------------------------------
// Shared WMMA GEMM main loop: one wave computes a 16(M) x 64(N) strip.
// ---------------------------------------------------------------------------
__device__ __forceinline__ void gemm_strip512(const _Float16* __restrict__ Arow,
                                              const _Float16* __restrict__ B,
                                              int n0, int m, int hh, v8f acc[4]) {
#pragma unroll 2
  for (int k0 = 0; k0 < 512; k0 += 32) {
    v16h a = load_afrag(Arow, k0, hh);
#pragma unroll
    for (int nt = 0; nt < 4; ++nt) {
      const _Float16* br = B + (size_t)(n0 + nt * 16 + m) * 512 + k0 + hh * 16;
      v16h bf = *(const v16h*)br;
      acc[nt] = WMMA_F16(a, bf, acc[nt]);
    }
  }
}

// ---------------------------------------------------------------------------
// Kernel 4: QKV projection + bias scatter into [b][n][q][64] f16 slabs.
// ---------------------------------------------------------------------------
__global__ __launch_bounds__(128) void gemm_qkv_kernel(
    const _Float16* __restrict__ A, const _Float16* __restrict__ B,
    const float* __restrict__ rwb, const float* __restrict__ rrb,
    _Float16* __restrict__ qw, _Float16* __restrict__ qrr,
    _Float16* __restrict__ kk, _Float16* __restrict__ vv) {
  const int wv = threadIdx.x >> 5, lane = threadIdx.x & 31;
  const int m = lane & 15, hh = lane >> 4;
  const int i0 = blockIdx.x * 64 + wv * 16;
  const int n0 = blockIdx.y * 64;
  v8f zero8 = {};
  v8f acc[4];
#pragma unroll
  for (int t = 0; t < 4; ++t) acc[t] = zero8;
  gemm_strip512(A + (size_t)(i0 + m) * 512, B, n0, m, hh, acc);
#pragma unroll
  for (int nt = 0; nt < 4; ++nt) {
#pragma unroll
    for (int v = 0; v < 8; ++v) {
      int row = i0 + v + 8 * hh;
      int col = n0 + nt * 16 + m;
      float val = acc[nt][v];
      int which = col >> 9;
      int rr = col & 511;
      int hn = rr >> 6, dh = rr & 63;
      int qi = row >> 2, bb = row & 3;
      size_t idx = (((size_t)bb * NHEAD + hn) * QLEN + qi) * DHEAD + dh;
      if (which == 0) {
        qw[idx]  = (_Float16)(val + rwb[hn * DHEAD + dh]);
        qrr[idx] = (_Float16)(val + rrb[hn * DHEAD + dh]);
      } else if (which == 1) {
        kk[idx] = (_Float16)val;
      } else {
        vv[idx] = (_Float16)val;
      }
    }
  }
}

// ---------------------------------------------------------------------------
// Kernel 5: r_k = r @ Wr^T  -> rk [n][c][64] f16
// ---------------------------------------------------------------------------
__global__ __launch_bounds__(128) void gemm_rk_kernel(const _Float16* __restrict__ A,
                                                      const _Float16* __restrict__ B,
                                                      _Float16* __restrict__ rkh) {
  const int wv = threadIdx.x >> 5, lane = threadIdx.x & 31;
  const int m = lane & 15, hh = lane >> 4;
  const int i0 = blockIdx.x * 64 + wv * 16;
  const int n0 = blockIdx.y * 64;
  v8f zero8 = {};
  v8f acc[4];
#pragma unroll
  for (int t = 0; t < 4; ++t) acc[t] = zero8;
  gemm_strip512(A + (size_t)(i0 + m) * 512, B, n0, m, hh, acc);
#pragma unroll
  for (int nt = 0; nt < 4; ++nt) {
#pragma unroll
    for (int v = 0; v < 8; ++v) {
      int row = i0 + v + 8 * hh;
      int col = n0 + nt * 16 + m;
      int hn = col >> 6, dh = col & 63;
      rkh[((size_t)hn * QLEN + row) * DHEAD + dh] = (_Float16)acc[nt][v];
    }
  }
}

// ---------------------------------------------------------------------------
// Kernel 6: fused rel-attention (flash-style online softmax).
// grid = (QLEN/64, BSZ*NHEAD), block = 128 (4 waves); wave owns 16 query rows.
// rel_shift: BDs[i,j] = qr_i . rk[q-1-i+j]   (j <= i)
//                     = 0                     (j == i+1)
//                     = qr_{i+1} . rk[j-i-2]  (j >= i+2)
// K tile is DMA'd into LDS by the Tensor Data Mover (with LDS padding to a
// 72-half row stride) when the tensor_load_to_lds builtin is available.
// ---------------------------------------------------------------------------
__global__ __launch_bounds__(128) void attn_kernel(
    const _Float16* __restrict__ qw, const _Float16* __restrict__ qrr,
    const _Float16* __restrict__ kk, const _Float16* __restrict__ vv,
    const _Float16* __restrict__ rk, _Float16* __restrict__ av) {
  __shared__ float    T1[64][132];
  __shared__ float    T2[64][132];
  __shared__ _Float16 Vt[64][72];      // transposed V tile: Vt[d][j]
  __shared__ _Float16 Kt[64][72];      // K tile (row-major, TDM-padded rows)
  __shared__ _Float16 Pj[4][16][72];   // per-wave P strip

  const int tid = threadIdx.x;
  const int wv = tid >> 5, lane = tid & 31;
  const int m = lane & 15, hh = lane >> 4;
  const int bn = blockIdx.y;
  const int n = bn & 7, b = bn >> 3;
  const int i0 = blockIdx.x * 64;

  const size_t slab = (size_t)bn * QLEN * DHEAD;
  const _Float16* qwS = qw + slab;
  const _Float16* qrS = qrr + slab;
  const _Float16* kkS = kk + slab;
  const _Float16* vvS = vv + slab;
  const _Float16* rkS = rk + (size_t)n * QLEN * DHEAD;

  // Query A-fragments (persist across the whole key loop)
  const int qrow = i0 + wv * 16 + m;
  int qrow1 = qrow + 1; if (qrow1 > QLEN - 1) qrow1 = QLEN - 1;
  v16h aQW[2], aQR[2], aQR1[2];
#pragma unroll
  for (int ks = 0; ks < 2; ++ks) {
    aQW[ks]  = load_afrag(qwS + (size_t)qrow  * DHEAD, ks * 32, hh);
    aQR[ks]  = load_afrag(qrS + (size_t)qrow  * DHEAD, ks * 32, hh);
    aQR1[ks] = load_afrag(qrS + (size_t)qrow1 * DHEAD, ks * 32, hh);
  }

  v8f zero8 = {};
  v16h zh;
#pragma unroll
  for (int t = 0; t < 16; ++t) zh[t] = (_Float16)0;

  v8f accO[4];
  float mrow[8], lrow[8];
#pragma unroll
  for (int t = 0; t < 4; ++t) accO[t] = zero8;
#pragma unroll
  for (int v = 0; v < 8; ++v) { mrow[v] = -3.0e38f; lrow[v] = 0.0f; }

#pragma unroll 1
  for (int j0 = 0; j0 < QLEN; j0 += 64) {
    __syncthreads();

    // Stage V tile transposed into LDS
    {
      int jl = tid & 63;
      int d0 = (tid >> 6) * 32;
      const _Float16* vrow = vvS + (size_t)(j0 + jl) * DHEAD + d0;
      v8h r0 = *(const v8h*)(vrow);
      v8h r1 = *(const v8h*)(vrow + 8);
      v8h r2 = *(const v8h*)(vrow + 16);
      v8h r3 = *(const v8h*)(vrow + 24);
#pragma unroll
      for (int t = 0; t < 8; ++t) {
        Vt[d0 + t     ][jl] = r0[t];
        Vt[d0 + t +  8][jl] = r1[t];
        Vt[d0 + t + 16][jl] = r2[t];
        Vt[d0 + t + 24][jl] = r3[t];
      }
    }

#if HAVE_TDM
    // Tensor Data Mover: DMA K tile [64 x 64] f16 -> Kt with 4-DWORD pad
    // every 32 DWORDs (row stride 72 halves).  Wave 0 issues, waits
    // TENSORcnt, barrier below publishes to the workgroup.
    if (wv == 0) {
      unsigned lds_off = (unsigned)(uintptr_t)(&Kt[0][0]);
      unsigned long long ga =
          (unsigned long long)(uintptr_t)(kkS + (size_t)j0 * DHEAD);
      u32x4 g0 = {
          1u,                                   // count=1, user mode
          lds_off,                              // lds_addr
          (unsigned)(ga & 0xffffffffu),         // global_addr[31:0]
          (unsigned)((ga >> 32) & 0x01ffffffu) | (2u << 30)  // addr hi | type=2
      };
      i32x8 g1 = {
          (int)((1u << 16) | (1u << 20) | (4u << 22) | (3u << 25)),
          // data_size=2B, pad_enable, pad_interval=32dw, pad_amount=4dw
          (int)(64u << 16),                     // tensor_dim0 = 64 (lo16)
          (int)(2048u << 16),                   // dim0 hi=0 | tensor_dim1 lo16
          (int)(64u << 16),                     // dim1 hi=0 | tile_dim0 = 64
          (int)64,                              // tile_dim1 = 64, tile_dim2 = 0
          (int)64,                              // tensor_dim0_stride = 64
          0, 0                                  // dim1_stride unused (2D tile)
      };
      i32x4 g2 = {0, 0, 0, 0};
      i32x4 g3 = {0, 0, 0, 0};
#if TDM_SIX_ARGS
      i32x8 g4 = {0, 0, 0, 0, 0, 0, 0, 0};
      __builtin_amdgcn_tensor_load_to_lds(g0, g1, g2, g3, g4, 0);
#else
      __builtin_amdgcn_tensor_load_to_lds(g0, g1, g2, g3, 0);
#endif
      __builtin_amdgcn_s_wait_tensorcnt(0);
    }
#endif

    // Prefetch next K tile into cache (global_prefetch_b8)
    if (j0 + 64 < QLEN)
      __builtin_prefetch(kkS + (size_t)(j0 + 64) * DHEAD + (size_t)tid * 32, 0, 1);

    // T1/T2 raw BD windows (wave's 16 rows x 128 window cols)
    const int c01 = QLEN - 64 - i0 + j0;  // T1 base: c = q-1-i+j = c01 + idx
    const int c02 = j0 - i0 - 65;         // T2 base: c = j-i-2   = c02 + idx
#pragma unroll 1
    for (int nt = 0; nt < 8; ++nt) {
      const int cc = nt * 16 + m;
      {
        int c = c01 + cc;
        bool ok = (c >= 0) && (c < QLEN);
        const _Float16* br = rkS + (size_t)(ok ? c : 0) * DHEAD;
        v8f acc = zero8;
#pragma unroll
        for (int ks = 0; ks < 2; ++ks) {
          v16h bf = ok ? *(const v16h*)(br + ks * 32 + hh * 16) : zh;
          acc = WMMA_F16(aQR[ks], bf, acc);
        }
#pragma unroll
        for (int v = 0; v < 8; ++v) T1[wv * 16 + v + 8 * hh][cc] = acc[v];
      }
      {
        int c = c02 + cc;
        bool ok = (c >= 0) && (c < QLEN);
        const _Float16* br = rkS + (size_t)(ok ? c : 0) * DHEAD;
        v8f acc = zero8;
#pragma unroll
        for (int ks = 0; ks < 2; ++ks) {
          v16h bf = ok ? *(const v16h*)(br + ks * 32 + hh * 16) : zh;
          acc = WMMA_F16(aQR1[ks], bf, acc);
        }
#pragma unroll
        for (int v = 0; v < 8; ++v) T2[wv * 16 + v + 8 * hh][cc] = acc[v];
      }
    }

    __syncthreads();   // Vt, Kt, T1, T2 all visible

    // AC = (q + r_w_bias) . K^T
    v8f accS[4];
#pragma unroll
    for (int t = 0; t < 4; ++t) accS[t] = zero8;
#pragma unroll 1
    for (int nt = 0; nt < 4; ++nt) {
#if HAVE_TDM
      const _Float16* br = &Kt[nt * 16 + m][0];
#pragma unroll
      for (int ks = 0; ks < 2; ++ks) {
        v8h blo = *(const v8h*)(br + ks * 32 + hh * 16);
        v8h bhi = *(const v8h*)(br + ks * 32 + hh * 16 + 8);
        accS[nt] = WMMA_F16(aQW[ks], pack16(blo, bhi), accS[nt]);
      }
#else
      const _Float16* br = kkS + (size_t)(j0 + nt * 16 + m) * DHEAD;
#pragma unroll
      for (int ks = 0; ks < 2; ++ks) {
        v16h bf = *(const v16h*)(br + ks * 32 + hh * 16);
        accS[nt] = WMMA_F16(aQW[ks], bf, accS[nt]);
      }
#endif
    }

    // S = (AC + BD_shifted) * scale   via diagonal gather from T1/T2
    float sv[4][8];
    const int liB = wv * 16 + 8 * hh;
#pragma unroll
    for (int nt = 0; nt < 4; ++nt) {
#pragma unroll
      for (int v = 0; v < 8; ++v) {
        int li = liB + v;
        int lj = nt * 16 + m;
        int dg = (j0 + lj) - (i0 + li);
        int idx = 63 + lj - li;
        float bd = (dg <= 0) ? T1[li][idx] : ((dg == 1) ? 0.0f : T2[li][idx]);
        sv[nt][v] = (accS[nt][v] + bd) * 0.125f;
      }
    }

    // Online softmax over the key dimension
    float alpha[8];
#pragma unroll
    for (int v = 0; v < 8; ++v) {
      float x = fmaxf(fmaxf(sv[0][v], sv[1][v]), fmaxf(sv[2][v], sv[3][v]));
#pragma unroll
      for (int msk = 1; msk < 16; msk <<= 1) x = fmaxf(x, __shfl_xor(x, msk, 32));
      float mn = fmaxf(mrow[v], x);
      alpha[v] = __expf(mrow[v] - mn);
      mrow[v] = mn;
    }
#pragma unroll
    for (int nt = 0; nt < 4; ++nt)
#pragma unroll
      for (int v = 0; v < 8; ++v) sv[nt][v] = __expf(sv[nt][v] - mrow[v]);
#pragma unroll
    for (int v = 0; v < 8; ++v) {
      float s = sv[0][v] + sv[1][v] + sv[2][v] + sv[3][v];
#pragma unroll
      for (int msk = 1; msk < 16; msk <<= 1) s += __shfl_xor(s, msk, 32);
      lrow[v] = lrow[v] * alpha[v] + s;
    }
#pragma unroll
    for (int nt = 0; nt < 4; ++nt)
#pragma unroll
      for (int v = 0; v < 8; ++v) accO[nt][v] *= alpha[v];
#pragma unroll
    for (int nt = 0; nt < 4; ++nt)
#pragma unroll
      for (int v = 0; v < 8; ++v)
        Pj[wv][v + 8 * hh][nt * 16 + m] = (_Float16)sv[nt][v];

    __syncthreads();

    // O += P @ V  (P from LDS as A-fragments, V^T from LDS as B-fragments)
    v16h aP[2];
#pragma unroll
    for (int ks = 0; ks < 2; ++ks) aP[ks] = load_afrag(&Pj[wv][m][0], ks * 32, hh);
#pragma unroll 1
    for (int nt = 0; nt < 4; ++nt) {
      const _Float16* vr = &Vt[nt * 16 + m][0];
#pragma unroll
      for (int ks = 0; ks < 2; ++ks) {
        v8h blo = *(const v8h*)(vr + ks * 32 + hh * 16);
        v8h bhi = *(const v8h*)(vr + ks * 32 + hh * 16 + 8);
        accO[nt] = WMMA_F16(aP[ks], pack16(blo, bhi), accO[nt]);
      }
    }
  }

  // Normalize and write attn_vec f16 [q*BSZ+b][n*64+d]
#pragma unroll
  for (int nt = 0; nt < 4; ++nt) {
#pragma unroll
    for (int v = 0; v < 8; ++v) {
      int rowq = i0 + wv * 16 + v + 8 * hh;
      int e = n * DHEAD + nt * 16 + m;
      av[((size_t)rowq * BSZ + b) * DMODEL + e] =
          (_Float16)(accO[nt][v] / lrow[v]);
    }
  }
}

// ---------------------------------------------------------------------------
// Kernel 7: out = w + attn_vec @ Wo^T   (f32 output)
// ---------------------------------------------------------------------------
__global__ __launch_bounds__(128) void gemm_out_kernel(const _Float16* __restrict__ A,
                                                       const _Float16* __restrict__ B,
                                                       const float* __restrict__ wres,
                                                       float* __restrict__ out) {
  const int wv = threadIdx.x >> 5, lane = threadIdx.x & 31;
  const int m = lane & 15, hh = lane >> 4;
  const int i0 = blockIdx.x * 64 + wv * 16;
  const int n0 = blockIdx.y * 64;
  v8f zero8 = {};
  v8f acc[4];
#pragma unroll
  for (int t = 0; t < 4; ++t) acc[t] = zero8;
  gemm_strip512(A + (size_t)(i0 + m) * 512, B, n0, m, hh, acc);
#pragma unroll
  for (int nt = 0; nt < 4; ++nt) {
#pragma unroll
    for (int v = 0; v < 8; ++v) {
      int row = i0 + v + 8 * hh;
      int col = n0 + nt * 16 + m;
      size_t o = (size_t)row * DMODEL + col;
      out[o] = wres[o] + acc[nt][v];
    }
  }
}

// ---------------------------------------------------------------------------
extern "C" void kernel_launch(void* const* d_in, const int* in_sizes, int n_in,
                              void* d_out, int out_size, void* d_ws, size_t ws_size,
                              hipStream_t stream) {
  (void)in_sizes; (void)n_in; (void)out_size; (void)ws_size;
  const float* w     = (const float*)d_in[0];
  const float* Wqkv  = (const float*)d_in[1];
  const float* Wr    = (const float*)d_in[2];
  const float* Wo    = (const float*)d_in[3];
  const float* gamma = (const float*)d_in[4];
  const float* beta  = (const float*)d_in[5];
  const float* rwb   = (const float*)d_in[6];
  const float* rrb   = (const float*)d_in[7];
  float* out = (float*)d_out;

  size_t off = 0;
  auto alloc = [&](size_t halves) -> _Float16* {
    _Float16* p = (_Float16*)((char*)d_ws + off);
    off += ((halves * sizeof(_Float16) + 255) / 256) * 256;
    return p;
  };
  const size_t ROWS = (size_t)QLEN * BSZ;               // 8192
  _Float16* wn    = alloc(ROWS * DMODEL);               // LN(w) f16
  _Float16* rh    = alloc((size_t)QLEN * DMODEL);       // pos-emb f16
  _Float16* wqkvh = alloc((size_t)3 * DMODEL * DMODEL); // Wqkv f16
  _Float16* wrh   = alloc((size_t)DMODEL * DMODEL);
  _Float16* woh   = alloc((size_t)DMODEL * DMODEL);
  const size_t SLAB = (size_t)BSZ * NHEAD * QLEN * DHEAD;
  _Float16* qwh = alloc(SLAB);                          // q + r_w_bias
  _Float16* qrh = alloc(SLAB);                          // q + r_r_bias
  _Float16* kkh = alloc(SLAB);
  _Float16* vvh = alloc(SLAB);
  _Float16* rkh = alloc((size_t)NHEAD * QLEN * DHEAD);
  _Float16* avh = alloc(ROWS * DMODEL);                 // attn_vec f16

  ln_cast_kernel<<<dim3((unsigned)ROWS), 128, 0, stream>>>(w, gamma, beta, wn);
  posemb_kernel<<<dim3(QLEN), 128, 0, stream>>>(rh);
  {
    int n1 = 3 * DMODEL * DMODEL, n2 = DMODEL * DMODEL;
    cast_f16_kernel<<<(n1 + 255) / 256, 256, 0, stream>>>(Wqkv, wqkvh, n1);
    cast_f16_kernel<<<(n2 + 255) / 256, 256, 0, stream>>>(Wr, wrh, n2);
    cast_f16_kernel<<<(n2 + 255) / 256, 256, 0, stream>>>(Wo, woh, n2);
  }
  gemm_qkv_kernel<<<dim3((unsigned)(ROWS / 64), (3 * DMODEL) / 64), 128, 0, stream>>>(
      wn, wqkvh, rwb, rrb, qwh, qrh, kkh, vvh);
  gemm_rk_kernel<<<dim3(QLEN / 64, DMODEL / 64), 128, 0, stream>>>(rh, wrh, rkh);
  attn_kernel<<<dim3(QLEN / 64, BSZ * NHEAD), 128, 0, stream>>>(qwh, qrh, kkh, vvh, rkh, avh);
  gemm_out_kernel<<<dim3((unsigned)(ROWS / 64), DMODEL / 64), 128, 0, stream>>>(
      avh, woh, w, out);
}
